// Speaker_4973572128948
// MI455X (gfx1250) — matrix operand order repaired and checked by reading, not compiled, not run on hardware
//
#include <hip/hip_runtime.h>
#include <hip/hip_bf16.h>

// ---------------------------------------------------------------------------
// Problem constants (from reference): N=16, Lt=Lv=1024, D=512, H=8, HD=64
// ---------------------------------------------------------------------------
#define NB   16
#define LT   1024
#define LV   1024
#define DM   512
#define NH   8
#define HD   64
#define DF   1024          // 2*D ffn width
#define MROWS (NB * LT)    // 16384

typedef __attribute__((ext_vector_type(16))) __bf16 v16bf;
typedef __attribute__((ext_vector_type(8)))  __bf16 v8bf;
typedef __attribute__((ext_vector_type(8)))  float  v8f;

struct Frag { union { v16bf v; v8bf h[2]; }; };

__device__ __forceinline__ v8f wmma_bf16(v16bf a, v16bf b, v8f c) {
    return __builtin_amdgcn_wmma_f32_16x16x32_bf16(
        false, a, false, b, (short)0, c, false, false);
}

__device__ __forceinline__ v8f splat8(float x) {
    v8f r;
#pragma unroll
    for (int i = 0; i < 8; ++i) r[i] = x;
    return r;
}
__device__ __forceinline__ v8f vmax8(v8f a, v8f b) {
    v8f r;
#pragma unroll
    for (int i = 0; i < 8; ++i) r[i] = fmaxf(a[i], b[i]);
    return r;
}
__device__ __forceinline__ v8f vexp8(v8f a) {
    v8f r;
#pragma unroll
    for (int i = 0; i < 8; ++i) r[i] = __expf(a[i]);
    return r;
}
__device__ __forceinline__ v8f shfl_xor8(v8f a, int m) {
    v8f r;
#pragma unroll
    for (int i = 0; i < 8; ++i) r[i] = __shfl_xor(a[i], m, 32);
    return r;
}

// ---------------------------------------------------------------------------
// Elementwise cast f32 -> bf16
// ---------------------------------------------------------------------------
__global__ void k_cast_bf16(const float* __restrict__ in,
                            __bf16* __restrict__ out, int n) {
    for (int i = blockIdx.x * blockDim.x + threadIdx.x; i < n;
         i += gridDim.x * blockDim.x)
        out[i] = (__bf16)in[i];
}

// Transposed cast: w (R,C) f32 -> wt (C,R) bf16
__global__ void k_cast_wT(const float* __restrict__ w,
                          __bf16* __restrict__ wt, int R, int C) {
    int n = R * C;
    for (int i = blockIdx.x * blockDim.x + threadIdx.x; i < n;
         i += gridDim.x * blockDim.x) {
        int c = i / R, r = i - c * R;
        wt[i] = (__bf16)w[(size_t)r * C + c];
    }
}

// V (N*L, D) bf16 -> Vt (N,H,HD,L) bf16  (per-head transposed for probs@V)
__global__ void k_transpose_v(const __bf16* __restrict__ v,
                              __bf16* __restrict__ vt, int L) {
    int n = NB * NH * HD * L;
    for (int i = blockIdx.x * blockDim.x + threadIdx.x; i < n;
         i += gridDim.x * blockDim.x) {
        int l  = i % L;
        int d  = (i / L) % HD;
        int h  = (i / (L * HD)) % NH;
        int b  = i / (L * HD * NH);
        vt[i] = v[((size_t)(b * L + l)) * DM + h * HD + d];
    }
}

// ---------------------------------------------------------------------------
// GEMM: C(M,N) = act(A(M,K) @ Bt(N,K)^T + bias)
// block = 256 threads (8 waves); WG tile M=256 (32 rows/wave), N=64,
// K-step 64, double-buffered LDS B tile (one barrier per stage).
// Per stage per wave: 4 A-frag loads, 16 LDS B-frag half-loads, 16 WMMAs.
// ---------------------------------------------------------------------------
__global__ __launch_bounds__(256)
void k_gemm(const __bf16* __restrict__ A, const __bf16* __restrict__ Bt,
            const float* __restrict__ bias,
            float* __restrict__ outF, __bf16* __restrict__ outB,
            int M, int N, int K, int gelu) {
    __shared__ __align__(32) __bf16 bs[2][64][72];

    const int lane = threadIdx.x & 31;
    const int wv   = threadIdx.x >> 5;
    const int m0   = blockIdx.y * 256 + wv * 32;
    const int n0   = blockIdx.x * 64;
    const int ncol = lane & 15;
    const int kof  = lane & 16;

    const __bf16* arow0 = A + (size_t)(m0 + ncol) * K + (kof >> 1);
    const __bf16* arow1 = arow0 + (size_t)16 * K;

    const int tr = threadIdx.x >> 2;        // 0..63
    const int tc = (threadIdx.x & 3) * 16;  // 0,16,32,48
    const __bf16* bsrc = Bt + (size_t)(n0 + tr) * K + tc;

    v8f acc0[4], acc1[4];
#pragma unroll
    for (int s = 0; s < 4; ++s) { acc0[s] = splat8(0.f); acc1[s] = splat8(0.f); }

    // prologue: stage tile 0 into buffer 0
    {
        v8bf r0 = *(const v8bf*)(bsrc);
        v8bf r1 = *(const v8bf*)(bsrc + 8);
        *(v8bf*)&bs[0][tr][tc]     = r0;
        *(v8bf*)&bs[0][tr][tc + 8] = r1;
    }
    __syncthreads();

    int buf = 0;
    for (int kk = 0; kk < K; kk += 64) {
        const bool haveNext = (kk + 64 < K);
        v8bf r0 = {}, r1 = {};
        if (haveNext) {
            r0 = *(const v8bf*)(bsrc + kk + 64);
            r1 = *(const v8bf*)(bsrc + kk + 64 + 8);
            __builtin_prefetch(arow0 + kk + 64, 0, 3);
            __builtin_prefetch(arow1 + kk + 64, 0, 3);
        }
#pragma unroll
        for (int ks = 0; ks < 2; ++ks) {
            const int kb = kk + ks * 32;
            Frag a0, a1;
            a0.h[0] = *(const v8bf*)(arow0 + kb);
            a0.h[1] = *(const v8bf*)(arow0 + kb + 16);
            a1.h[0] = *(const v8bf*)(arow1 + kb);
            a1.h[1] = *(const v8bf*)(arow1 + kb + 16);
#pragma unroll
            for (int sub = 0; sub < 4; ++sub) {
                Frag bfg;
                const __bf16* bp = &bs[buf][sub * 16 + ncol][ks * 32 + (kof >> 1)];
                bfg.h[0] = *(const v8bf*)(bp);
                bfg.h[1] = *(const v8bf*)(bp + 16);
                acc0[sub] = wmma_bf16(a0.v, bfg.v, acc0[sub]);
                acc1[sub] = wmma_bf16(a1.v, bfg.v, acc1[sub]);
            }
        }
        if (haveNext) {
            *(v8bf*)&bs[buf ^ 1][tr][tc]     = r0;
            *(v8bf*)&bs[buf ^ 1][tr][tc + 8] = r1;
        }
        __syncthreads();
        buf ^= 1;
    }

    const int mbase = (lane >> 4) << 3;
#pragma unroll
    for (int rt = 0; rt < 2; ++rt) {
        const v8f* acc = rt ? acc1 : acc0;
        const int mrow = m0 + rt * 16 + mbase;
#pragma unroll
        for (int sub = 0; sub < 4; ++sub) {
            int col = n0 + sub * 16 + ncol;
            float bv = bias ? bias[col] : 0.f;
#pragma unroll
            for (int j = 0; j < 8; ++j) {
                float val = acc[sub][j] + bv;
                if (gelu) val = 0.5f * val * (1.f + erff(val * 0.70710678f));
                size_t o = (size_t)(mrow + j) * N + col;
                if (outF) outF[o] = val;
                if (outB) outB[o] = (__bf16)val;
            }
        }
    }
}

// ---------------------------------------------------------------------------
// Flash attention.  block = 128 threads (4 waves), each wave owns a 16-row
// q-tile; grid = (Lq/64, H, N).  Processes keys 64 at a time:
//   8 WMMA (scores) + one online-softmax reduction + LDS relayout +
//   8 WMMA (probs@V).
// IS_SELF: causal+padding additive bias.  !IS_SELF: raw scaled scores are
// streamed to `vis` (the attention_vis output).
// ---------------------------------------------------------------------------
template <bool IS_SELF>
__global__ __launch_bounds__(128)
void k_attn(const __bf16* __restrict__ Qb, const __bf16* __restrict__ Kb,
            const __bf16* __restrict__ Vt, const float* __restrict__ mask,
            float* __restrict__ ctx, float* __restrict__ vis,
            int Lq, int Lk) {
    __shared__ __align__(32) __bf16 plds[4][16][72];

    const int lane = threadIdx.x & 31;
    const int wv   = threadIdx.x >> 5;
    const int b    = blockIdx.z;
    const int h    = blockIdx.y;
    const int q0   = blockIdx.x * 64 + wv * 16;

    const __bf16* qbase = Qb + (size_t)(b * Lq + q0) * DM + h * HD;
    const __bf16* kbase = Kb + (size_t)b * Lk * DM + h * HD;
    const __bf16* vbase = Vt + ((size_t)(b * NH + h)) * HD * Lk;

    const int mrow = (lane >> 4) << 3;          // row slot base (0 or 8)
    const int ncol = lane & 15;
    const int dofs = lane & 16;

    Frag qf[2];
    {
        const __bf16* qr = qbase + (size_t)ncol * DM + (dofs >> 1);
        qf[0].h[0] = *(const v8bf*)(qr);
        qf[0].h[1] = *(const v8bf*)(qr + 16);
        qf[1].h[0] = *(const v8bf*)(qr + 32);
        qf[1].h[1] = *(const v8bf*)(qr + 48);
    }

    v8f acc[4];
#pragma unroll
    for (int s = 0; s < 4; ++s) acc[s] = splat8(0.f);
    v8f m_run = splat8(-1e30f);
    v8f l_run = splat8(0.f);

    for (int kt = 0; kt < Lk / 64; ++kt) {
        v8f s[4];
#pragma unroll
        for (int t = 0; t < 4; ++t) {
            int kb = kt * 64 + t * 16;
            const __bf16* kp = kbase + (size_t)(kb + ncol) * DM;
            Frag b0, b1;
            b0.h[0] = *(const v8bf*)(kp + (dofs >> 1));
            b0.h[1] = *(const v8bf*)(kp + (dofs >> 1) + 16);
            b1.h[0] = *(const v8bf*)(kp + 32 + (dofs >> 1));
            b1.h[1] = *(const v8bf*)(kp + 48 + (dofs >> 1));

            v8f sc = splat8(0.f);
            sc = wmma_bf16(qf[0].v, b0.v, sc);
            sc = wmma_bf16(qf[1].v, b1.v, sc);
            sc = sc * 0.125f;                   // 1/sqrt(HD)

            if (IS_SELF) {
                int kg = kb + ncol;
                float mv = mask[b * Lk + kg];
#pragma unroll
                for (int j = 0; j < 8; ++j) {
                    int qg = q0 + mrow + j;
                    float keep = (kg <= qg) ? mv : 0.f;
                    sc[j] += (1.f - keep) * -10000.f;
                }
            } else {
                float* vp = vis +
                    (((size_t)(b * NH + h) * Lq + (q0 + mrow)) * Lk) + kb + ncol;
#pragma unroll
                for (int j = 0; j < 8; ++j)
                    vp[(size_t)j * Lk] = sc[j];
            }
            s[t] = sc;
        }

        // ---- single online-softmax update for all 64 keys ----------------
        v8f t = vmax8(vmax8(s[0], s[1]), vmax8(s[2], s[3]));
#pragma unroll
        for (int off = 1; off < 16; off <<= 1) t = vmax8(t, shfl_xor8(t, off));
        v8f m_new = vmax8(m_run, t);
        v8f alpha = vexp8(m_run - m_new);
#pragma unroll
        for (int i = 0; i < 4; ++i) s[i] = vexp8(s[i] - m_new);
        v8f ls = (s[0] + s[1]) + (s[2] + s[3]);
#pragma unroll
        for (int off = 1; off < 16; off <<= 1) ls = ls + shfl_xor8(ls, off);
        l_run = l_run * alpha + ls;
        m_run = m_new;
#pragma unroll
        for (int d = 0; d < 4; ++d) acc[d] = acc[d] * alpha;

        // ---- relayout probs (C-layout -> A-fragments) via per-wave LDS ---
#pragma unroll
        for (int t2 = 0; t2 < 4; ++t2)
#pragma unroll
            for (int j = 0; j < 8; ++j)
                plds[wv][mrow + j][t2 * 16 + ncol] = (__bf16)s[t2][j];

        Frag pf0, pf1;
        {
            const __bf16* pr = &plds[wv][ncol][dofs >> 1];
            pf0.h[0] = *(const v8bf*)(pr);
            pf0.h[1] = *(const v8bf*)(pr + 16);
            pf1.h[0] = *(const v8bf*)(pr + 32);
            pf1.h[1] = *(const v8bf*)(pr + 48);
        }

        // ---- ctx += P @ V ------------------------------------------------
#pragma unroll
        for (int d = 0; d < 4; ++d) {
            const __bf16* vp =
                vbase + (size_t)(d * 16 + ncol) * Lk + kt * 64 + (dofs >> 1);
            Frag vf0, vf1;
            vf0.h[0] = *(const v8bf*)(vp);
            vf0.h[1] = *(const v8bf*)(vp + 16);
            vf1.h[0] = *(const v8bf*)(vp + 32);
            vf1.h[1] = *(const v8bf*)(vp + 48);
            acc[d] = wmma_bf16(pf0.v, vf0.v, acc[d]);
            acc[d] = wmma_bf16(pf1.v, vf1.v, acc[d]);
        }
    }

    v8f inv;
#pragma unroll
    for (int j = 0; j < 8; ++j) inv[j] = 1.f / fmaxf(l_run[j], 1e-30f);
#pragma unroll
    for (int d = 0; d < 4; ++d)
#pragma unroll
        for (int j = 0; j < 8; ++j)
            ctx[(size_t)(b * Lq + q0 + mrow + j) * DM + h * HD + d * 16 + ncol] =
                acc[d][j] * inv[j];
}

// ---------------------------------------------------------------------------
// Fused residual + LayerNorm: y = LN(a+b)*g + beta.  One wave per 512-row.
// ---------------------------------------------------------------------------
__global__ __launch_bounds__(128)
void k_res_ln(const float* __restrict__ a, const float* __restrict__ bsrc,
              const float* __restrict__ g, const float* __restrict__ be,
              float* __restrict__ outF, __bf16* __restrict__ outB) {
    const int lane = threadIdx.x & 31;
    const int row  = blockIdx.x * 4 + (threadIdx.x >> 5);
    const float* pa = a + (size_t)row * DM;
    const float* pb = bsrc + (size_t)row * DM;

    float x[16], s = 0.f, s2 = 0.f;
#pragma unroll
    for (int i = 0; i < 16; ++i) {
        int c = lane + i * 32;
        float v = pa[c] + pb[c];
        x[i] = v; s += v; s2 += v * v;
    }
#pragma unroll
    for (int off = 1; off < 32; off <<= 1) {
        s  += __shfl_xor(s, off, 32);
        s2 += __shfl_xor(s2, off, 32);
    }
    float mu = s * (1.f / DM);
    float var = s2 * (1.f / DM) - mu * mu;
    float rs = rsqrtf(var + 1e-6f);
#pragma unroll
    for (int i = 0; i < 16; ++i) {
        int c = lane + i * 32;
        float y = (x[i] - mu) * rs * g[c] + be[c];
        if (outF) outF[(size_t)row * DM + c] = y;
        if (outB) outB[(size_t)row * DM + c] = (__bf16)y;
    }
}

// ---------------------------------------------------------------------------
// Host launcher
// ---------------------------------------------------------------------------
extern "C" void kernel_launch(void* const* d_in, const int* in_sizes, int n_in,
                              void* d_out, int out_size, void* d_ws,
                              size_t ws_size, hipStream_t stream) {
    (void)in_sizes; (void)n_in; (void)out_size; (void)ws_size;

    const float* dec  = (const float*)d_in[0];
    const float* dmsk = (const float*)d_in[1];
    const float* enc  = (const float*)d_in[2];
    const float* sa_qw = (const float*)d_in[3];  const float* sa_qb = (const float*)d_in[4];
    const float* sa_kw = (const float*)d_in[5];  const float* sa_kb = (const float*)d_in[6];
    const float* sa_vw = (const float*)d_in[7];  const float* sa_vb = (const float*)d_in[8];
    const float* n1_g  = (const float*)d_in[9];  const float* n1_b  = (const float*)d_in[10];
    const float* ca_qw = (const float*)d_in[11]; const float* ca_qb = (const float*)d_in[12];
    const float* ca_kw = (const float*)d_in[13]; const float* ca_kb = (const float*)d_in[14];
    const float* ca_vw = (const float*)d_in[15]; const float* ca_vb = (const float*)d_in[16];
    const float* n2_g  = (const float*)d_in[17]; const float* n2_b  = (const float*)d_in[18];
    const float* fc1_w = (const float*)d_in[19]; const float* fc1_b = (const float*)d_in[20];
    const float* fc2_w = (const float*)d_in[21]; const float* fc2_b = (const float*)d_in[22];
    const float* n3_g  = (const float*)d_in[23]; const float* n3_b  = (const float*)d_in[24];

    char* ws = (char*)d_ws;
    size_t off = 0;
    auto alloc = [&](size_t bytes) -> void* {
        void* p = ws + off;
        off += (bytes + 255) & ~(size_t)255;
        return p;
    };

    const size_t MD = (size_t)MROWS * DM;
    // Reused region: xb (first MD halfs) during projections, later h (M x DF).
    __bf16* hreg = (__bf16*)alloc((size_t)MROWS * DF * 2);
    __bf16* xb   = hreg;
    __bf16* eb   = (__bf16*)alloc(MD * 2);
    __bf16* wqs  = (__bf16*)alloc((size_t)DM * DM * 2);
    __bf16* wks  = (__bf16*)alloc((size_t)DM * DM * 2);
    __bf16* wvs  = (__bf16*)alloc((size_t)DM * DM * 2);
    __bf16* wqc  = (__bf16*)alloc((size_t)DM * DM * 2);
    __bf16* wkc  = (__bf16*)alloc((size_t)DM * DM * 2);
    __bf16* wvc  = (__bf16*)alloc((size_t)DM * DM * 2);
    __bf16* w1t  = (__bf16*)alloc((size_t)DF * DM * 2);  // (DF, D)
    __bf16* w2t  = (__bf16*)alloc((size_t)DM * DF * 2);  // (D, DF)
    __bf16* Qb   = (__bf16*)alloc(MD * 2);
    __bf16* Kb   = (__bf16*)alloc(MD * 2);
    __bf16* Vb   = (__bf16*)alloc(MD * 2);
    __bf16* Vt   = (__bf16*)alloc(MD * 2);     // N*H*HD*L == M*D
    float*  ctxF = (float*)alloc(MD * 4);      // reused 3x
    float*  attF = (float*)alloc(MD * 4);
    __bf16* attB = (__bf16*)alloc(MD * 2);
    float*  xF   = (float*)alloc(MD * 4);
    __bf16* xB   = (__bf16*)alloc(MD * 2);

    float* outMain = (float*)d_out;
    float* vis     = (float*)d_out + MD;

    const dim3 B256(256), B128(128);
    const int  GS = 2048;   // grid-stride grid for elementwise kernels

    // ---- casts -----------------------------------------------------------
    k_cast_bf16<<<GS, B256, 0, stream>>>(dec, xb, (int)MD);
    k_cast_bf16<<<GS, B256, 0, stream>>>(enc, eb, (int)MD);
    k_cast_wT<<<512, B256, 0, stream>>>(sa_qw, wqs, DM, DM);
    k_cast_wT<<<512, B256, 0, stream>>>(sa_kw, wks, DM, DM);
    k_cast_wT<<<512, B256, 0, stream>>>(sa_vw, wvs, DM, DM);
    k_cast_wT<<<512, B256, 0, stream>>>(ca_qw, wqc, DM, DM);
    k_cast_wT<<<512, B256, 0, stream>>>(ca_kw, wkc, DM, DM);
    k_cast_wT<<<512, B256, 0, stream>>>(ca_vw, wvc, DM, DM);
    k_cast_wT<<<1024, B256, 0, stream>>>(fc1_w, w1t, DM, DF);
    k_cast_wT<<<1024, B256, 0, stream>>>(fc2_w, w2t, DF, DM);

    // ---- self-attention QKV projections ---------------------------------
    dim3 gP(DM / 64, MROWS / 256);
    k_gemm<<<gP, B256, 0, stream>>>(xb, wqs, sa_qb, nullptr, Qb, MROWS, DM, DM, 0);
    k_gemm<<<gP, B256, 0, stream>>>(xb, wks, sa_kb, nullptr, Kb, MROWS, DM, DM, 0);
    k_gemm<<<gP, B256, 0, stream>>>(xb, wvs, sa_vb, nullptr, Vb, MROWS, DM, DM, 0);
    k_transpose_v<<<GS, B256, 0, stream>>>(Vb, Vt, LT);

    // ---- self-attention (flash, causal+pad bias) -> ctxF -----------------
    dim3 gA(LT / 64, NH, NB);
    k_attn<true><<<gA, B128, 0, stream>>>(Qb, Kb, Vt, dmsk, ctxF, nullptr, LT, LT);

    // ---- residual + LN1 --------------------------------------------------
    k_res_ln<<<MROWS / 4, B128, 0, stream>>>(dec, ctxF, n1_g, n1_b, attF, attB);

    // ---- cross-attention projections (reuse Qb/Kb/Vb/Vt) -----------------
    k_gemm<<<gP, B256, 0, stream>>>(attB, wqc, ca_qb, nullptr, Qb, MROWS, DM, DM, 0);
    k_gemm<<<gP, B256, 0, stream>>>(eb, wkc, ca_kb, nullptr, Kb, MROWS, DM, DM, 0);
    k_gemm<<<gP, B256, 0, stream>>>(eb, wvc, ca_vb, nullptr, Vb, MROWS, DM, DM, 0);
    k_transpose_v<<<GS, B256, 0, stream>>>(Vb, Vt, LV);

    // ---- cross-attention: vis scores -> d_out, ctx -> ctxF ---------------
    k_attn<false><<<gA, B128, 0, stream>>>(Qb, Kb, Vt, nullptr, ctxF, vis, LT, LV);

    // ---- residual + LN2 --------------------------------------------------
    k_res_ln<<<MROWS / 4, B128, 0, stream>>>(attF, ctxF, n2_g, n2_b, xF, xB);

    // ---- FFN -------------------------------------------------------------
    dim3 gF1(DF / 64, MROWS / 256);
    k_gemm<<<gF1, B256, 0, stream>>>(xB, w1t, fc1_b, nullptr, hreg, MROWS, DF, DM, 1);
    k_gemm<<<gP, B256, 0, stream>>>(hreg, w2t, fc2_b, ctxF, nullptr, MROWS, DM, DF, 0);

    // ---- residual + LN3 -> main output -----------------------------------
    k_res_ln<<<MROWS / 4, B128, 0, stream>>>(xF, ctxF, n3_g, n3_b, outMain, nullptr);
}
